// GNNModel_16346645528761
// MI455X (gfx1250) — compile-verified, matrix operand order checked by reading, not run on hardware
//
#include <hip/hip_runtime.h>
#include <hip/hip_bf16.h>
#include <stdint.h>

// ---------------------------------------------------------------------------
// Types for CDNA5 WMMA (wave32): v_wmma_f32_16x16x32_bf16
// ---------------------------------------------------------------------------
typedef __bf16  v16bf __attribute__((ext_vector_type(16)));
typedef __bf16  v8bf  __attribute__((ext_vector_type(8)));
typedef float   v8f   __attribute__((ext_vector_type(8)));

#define HS   128
#define HD   256
#define PADL 512

static inline unsigned cdivz(size_t a, int b) { return (unsigned)((a + (size_t)b - 1) / (size_t)b); }

// ---------------------------------------------------------------------------
// ordered-uint encoding of float for atomic max (monotone, handles negatives)
// ---------------------------------------------------------------------------
__device__ __forceinline__ unsigned f2ord(float f) {
    unsigned b = __float_as_uint(f);
    return (b & 0x80000000u) ? ~b : (b | 0x80000000u);
}
__device__ __forceinline__ float ord2f(unsigned o) {
    unsigned b = (o & 0x80000000u) ? (o & 0x7fffffffu) : ~o;
    return __uint_as_float(b);
}
// f2ord(-inf) = ~0xFF800000 = 0x007FFFFF

// ---------------------------------------------------------------------------
// Utility kernels
// ---------------------------------------------------------------------------
__global__ void k_fill_u32(unsigned* p, unsigned v, size_t n) {
    size_t i = (size_t)blockIdx.x * blockDim.x + threadIdx.x;
    if (i < n) p[i] = v;
}

__global__ void k_cast_bf16(const float* __restrict__ in, __bf16* __restrict__ out, size_t n) {
    size_t i = (size_t)blockIdx.x * blockDim.x + threadIdx.x;
    if (i < n) out[i] = (__bf16)in[i];
}

// Bt[n*K + k] = (bf16) W[k*N + n]
__global__ void k_transpose_cast(const float* __restrict__ W, __bf16* __restrict__ Bt, int K, int Nc) {
    int i = blockIdx.x * blockDim.x + threadIdx.x;
    if (i >= K * Nc) return;
    int k = i / Nc, n = i % Nc;
    Bt[(size_t)n * K + k] = (__bf16)W[i];
}

// x_g gather: abf[node*128+c] = bf16(emb_tab_g[x[node]*128+c])
__global__ void k_gather_g(const int* __restrict__ x, const float* __restrict__ tabg,
                           __bf16* __restrict__ out, int Nn) {
    int i = blockIdx.x * blockDim.x + threadIdx.x;
    if (i >= Nn * HS) return;
    int node = i >> 7, c = i & 127;
    out[i] = (__bf16)tabg[x[node] * HS + c];
}

// ---------------------------------------------------------------------------
// WMMA bf16 GEMM: C[M,N] = A[M,K] * Bt[N,K]^T (+bias)(+relu)
// Register-blocked 1x4: one wave owns a 16x64 C strip (4 accumulators),
// reusing each A fragment across 4 B fragments -> 4 wmma per 10 x 16B loads.
// (M-tile, N64-tile) space is flattened across waves so narrow-N GEMMs keep
// all 8 waves of the block busy.
// A-fragment layout (16-bit A 16x32): lane l (row M = l&15) holds
//   K = {0..7,16..23} (l<16) or {8..15,24..31} (l>=16)  -> two 16B loads.
// B stored N-major so B-fragment addressing is identical.
// C layout: VGPR r, lane l -> M = r + (l>=16 ? 8:0), N = l&15.
// ---------------------------------------------------------------------------
#define GEMM_WAVES 8
#define NB 4
__global__ __launch_bounds__(256) void k_gemm_bf16(
    const __bf16* __restrict__ A, const __bf16* __restrict__ Bt,
    const float* __restrict__ bias, float* __restrict__ C,
    int M, int Ncols, int K, int relu)
{
    int lane  = threadIdx.x & 31;
    int wave  = threadIdx.x >> 5;
    int nT64  = Ncols >> 6;                    // 64-col tiles (all N are mult of 64)
    int tiles = nT64 * (M >> 4);
    int wg    = blockIdx.x * GEMM_WAVES + wave;
    if (wg >= tiles) return;                   // wave-uniform: EXEC stays all-ones
    int tileM   = wg / nT64;
    int tileN64 = wg % nT64;

    int half = (lane >> 4) & 1;
    int mr   = lane & 15;
    int sel  = half * 8;
    const __bf16* ap = A  + (size_t)(tileM * 16 + mr) * K;
    const __bf16* bp = Bt + (size_t)(tileN64 * 64 + mr) * K;
    const size_t bstep = (size_t)16 * K;       // 16 columns of Bt

    v8f acc[NB] = {};
    for (int k0 = 0; k0 < K; k0 += 32) {
        v8bf alo = *(const v8bf*)(ap + k0 + sel);
        v8bf ahi = *(const v8bf*)(ap + k0 + 16 + sel);
        __builtin_prefetch(ap + k0 + 128, 0, 1);            // global_prefetch_b8
        v16bf a = __builtin_shufflevector(alo, ahi, 0,1,2,3,4,5,6,7,8,9,10,11,12,13,14,15);
        #pragma unroll
        for (int j = 0; j < NB; ++j) {
            const __bf16* bj = bp + bstep * j;
            v8bf blo = *(const v8bf*)(bj + k0 + sel);
            v8bf bhi = *(const v8bf*)(bj + k0 + 16 + sel);
            v16bf b = __builtin_shufflevector(blo, bhi, 0,1,2,3,4,5,6,7,8,9,10,11,12,13,14,15);
            acc[j] = __builtin_amdgcn_wmma_f32_16x16x32_bf16(
                         false, a, false, b, (short)0, acc[j], false, false);
        }
    }

    int m0 = tileM * 16 + half * 8;
    #pragma unroll
    for (int j = 0; j < NB; ++j) {
        int col = tileN64 * 64 + j * 16 + mr;
        float bv = bias ? bias[col] : 0.0f;
        #pragma unroll
        for (int r = 0; r < 8; ++r) {
            float v = acc[j][r] + bv;
            if (relu) v = fmaxf(v, 0.0f);
            C[(size_t)(m0 + r) * Ncols + col] = v;
        }
    }
}

// ---------------------------------------------------------------------------
// GAT attention pipeline
// ---------------------------------------------------------------------------
__global__ void k_attn_coeff(const float* __restrict__ h, const float* __restrict__ aw_s,
                             const float* __restrict__ aw_d, float* __restrict__ As,
                             float* __restrict__ Ad, int Nn, int H, int C) {
    int i = blockIdx.x * blockDim.x + threadIdx.x;
    if (i >= Nn * H) return;
    int n = i / H, hh = i % H;
    const float* hp = h + (size_t)n * H * C + (size_t)hh * C;
    const float* sp = aw_s + hh * C;
    const float* dp = aw_d + hh * C;
    float s = 0.f, d = 0.f;
    for (int c = 0; c < C; c += 4) {
        float4 v = *(const float4*)(hp + c);
        float4 ws = *(const float4*)(sp + c);
        float4 wd = *(const float4*)(dp + c);
        s += v.x * ws.x + v.y * ws.y + v.z * ws.z + v.w * ws.w;
        d += v.x * wd.x + v.y * wd.y + v.z * wd.z + v.w * wd.w;
    }
    As[i] = s; Ad[i] = d;
}

__global__ void k_edge_logit(const int* __restrict__ src, const int* __restrict__ dst,
                             int E, int Etot, const float* __restrict__ As,
                             const float* __restrict__ Ad, float* __restrict__ ebuf,
                             unsigned* __restrict__ mord, int H) {
    size_t i = (size_t)blockIdx.x * blockDim.x + threadIdx.x;
    if (i >= (size_t)Etot * H) return;
    int e = (int)(i / H), hh = (int)(i % H);
    int s = (e < E) ? src[e] : (e - E);
    int d = (e < E) ? dst[e] : (e - E);
    float v = As[s * H + hh] + Ad[d * H + hh];
    v = (v > 0.f) ? v : 0.2f * v;               // leaky_relu(0.2)
    ebuf[i] = v;
    atomicMax(mord + d * H + hh, f2ord(v));
}

__global__ void k_edge_exp(const int* __restrict__ dst, int E, int Etot,
                           float* __restrict__ ebuf, const unsigned* __restrict__ mord,
                           float* __restrict__ den, int H) {
    size_t i = (size_t)blockIdx.x * blockDim.x + threadIdx.x;
    if (i >= (size_t)Etot * H) return;
    int e = (int)(i / H), hh = (int)(i % H);
    int d = (e < E) ? dst[e] : (e - E);
    float m = ord2f(mord[d * H + hh]);          // finite: every node has a self loop
    float ex = __expf(ebuf[i] - m);
    ebuf[i] = ex;
    atomicAdd(den + d * H + hh, ex);
}

__global__ void k_edge_agg(const int* __restrict__ src, const int* __restrict__ dst,
                           int E, int Etot, const float* __restrict__ ebuf,
                           const float* __restrict__ den, const float* __restrict__ h,
                           float* __restrict__ out, int H, int C) {
    size_t i = (size_t)blockIdx.x * blockDim.x + threadIdx.x;
    int c4 = C / 4;
    size_t total = (size_t)Etot * H * c4;
    if (i >= total) return;
    int hc = (int)(i % c4);
    size_t rest = i / c4;
    int hh = (int)(rest % H);
    int e  = (int)(rest / H);
    int s = (e < E) ? src[e] : (e - E);
    int d = (e < E) ? dst[e] : (e - E);
    float ex = ebuf[(size_t)e * H + hh];
    float dn = den[d * H + hh];
    float alpha = ex / fmaxf(dn, 1e-16f);
    const float4 hv = *(const float4*)(h + (size_t)s * H * C + (size_t)hh * C + hc * 4);
    float* op = out + (size_t)d * H * C + (size_t)hh * C + hc * 4;
    atomicAdd(op + 0, alpha * hv.x);
    atomicAdd(op + 1, alpha * hv.y);
    atomicAdd(op + 2, alpha * hv.z);
    atomicAdd(op + 3, alpha * hv.w);
}

// mean over heads + bias + relu; optional bf16 and f32 outputs
__global__ void k_head_mean(const float* __restrict__ agg, const float* __restrict__ bias,
                            int Nn, int H, int C, __bf16* bfout, float* fout) {
    size_t i = (size_t)blockIdx.x * blockDim.x + threadIdx.x;
    if (i >= (size_t)Nn * C) return;
    int n = (int)(i / C), c = (int)(i % C);
    float s = 0.f;
    for (int hh = 0; hh < H; ++hh) s += agg[(size_t)n * H * C + (size_t)hh * C + c];
    float v = fmaxf(s * (1.0f / H) + bias[c], 0.f);
    if (bfout) bfout[i] = (__bf16)v;
    if (fout)  fout[i]  = v;
}

// ---------------------------------------------------------------------------
// Pooling / padding / conv / tail
// ---------------------------------------------------------------------------
__global__ void k_offsets(const int* __restrict__ node_len, int* __restrict__ offs, int Bn) {
    if (blockIdx.x == 0 && threadIdx.x == 0) {
        int a = 0;
        for (int b = 0; b < Bn; ++b) { offs[b] = a; a += node_len[b]; }
        offs[Bn] = a;
    }
}

__global__ void k_graph_pool(const float* __restrict__ hfin, const int* __restrict__ offs,
                             const int* __restrict__ node_len, int Bn, float* __restrict__ out) {
    int i = blockIdx.x * blockDim.x + threadIdx.x;
    if (i >= Bn * HS) return;
    int b = i / HS, c = i % HS;
    int off = offs[b], len = node_len[b];
    float s = 0.f;
    for (int r = 0; r < len; ++r) s += hfin[(size_t)(off + r) * HS + c];
    out[i] = s / (float)len;
}

__global__ void k_mask(const int* __restrict__ node_len, float* m1, float* m2, int Bn) {
    int i = blockIdx.x * blockDim.x + threadIdx.x;
    if (i >= Bn * PADL) return;
    int b = i / PADL, p = i % PADL;
    float v = (p < node_len[b]) ? 1.0f : 0.0f;
    m1[i] = v; m2[i] = v;
}

__global__ void k_outgraph(const float* __restrict__ hfin, const int* __restrict__ offs,
                           const int* __restrict__ node_len, int Nn, int Bn,
                           float* __restrict__ out) {
    size_t i = (size_t)blockIdx.x * blockDim.x + threadIdx.x;
    if (i >= (size_t)Bn * PADL * HS) return;
    int c = (int)(i % HS);
    size_t rest = i / HS;
    int p = (int)(rest % PADL);
    int b = (int)(rest / PADL);
    float v = 0.f;
    if (p < node_len[b]) {
        int gi = offs[b] + p; if (gi > Nn - 1) gi = Nn - 1;
        v = hfin[(size_t)gi * HS + c];
    }
    out[i] = v;
}

// xt[b][c][p] = valid ? 0.5*(emb_tab_r[x[gi]][c] + emb2[gi][c]) : 0
__global__ void k_build_xt(const int* __restrict__ x, const float* __restrict__ tabr,
                           const float* __restrict__ emb2, const int* __restrict__ offs,
                           const int* __restrict__ node_len, int Nn, int Bn,
                           float* __restrict__ xt) {
    size_t i = (size_t)blockIdx.x * blockDim.x + threadIdx.x;
    if (i >= (size_t)Bn * HS * PADL) return;
    int p = (int)(i % PADL);
    size_t rest = i / PADL;
    int c = (int)(rest % HS);
    int b = (int)(rest / HS);
    float v = 0.f;
    if (p < node_len[b]) {
        int gi = offs[b] + p; if (gi > Nn - 1) gi = Nn - 1;
        v = 0.5f * (tabr[x[gi] * HS + c] + emb2[(size_t)gi * HS + c]);
    }
    xt[i] = v;
}

// combined 15-tap weights: wcomb = (cw3 + pad(cw2) + pad(cw1)) / 3
__global__ void k_wcomb(const float* __restrict__ cw1, const float* __restrict__ cw2,
                        const float* __restrict__ cw3, float* __restrict__ wcomb) {
    int i = blockIdx.x * blockDim.x + threadIdx.x;
    if (i >= 64 * 128 * 15) return;
    int t = i % 15;
    int rest = i / 15;
    int ic = rest % 128, oc = rest / 128;
    float a = cw3[i];                                                // 15-tap
    if (t >= 2 && t <= 12) a += cw2[(oc * 128 + ic) * 11 + (t - 2)]; // 11-tap
    if (t >= 4 && t <= 10) a += cw1[(oc * 128 + ic) * 7 + (t - 4)];  // 7-tap
    wcomb[i] = a * (1.0f / 3.0f);
}

// direct fused conv, LDS-tiled input. y[(b*512+p)*64 + oc]
#define PTILE 32
__global__ __launch_bounds__(256) void k_conv(const float* __restrict__ xt,
                                              const float* __restrict__ wcomb,
                                              const float* __restrict__ cb1,
                                              const float* __restrict__ cb2,
                                              const float* __restrict__ cb3,
                                              float* __restrict__ y) {
    __shared__ float tile[128 * (PTILE + 14)];
    const int W = PTILE + 14;
    int b  = blockIdx.x;
    int p0 = blockIdx.y * PTILE;
    const float* xb = xt + (size_t)b * HS * PADL;
    for (int t = threadIdx.x; t < 128 * W; t += 256) {
        int ic = t / W, pp = t % W;
        int gp = p0 + pp - 7;
        tile[t] = (gp >= 0 && gp < PADL) ? xb[(size_t)ic * PADL + gp] : 0.f;
    }
    __syncthreads();
    int oc = threadIdx.x & 63;
    float bias = (cb1[oc] + cb2[oc] + cb3[oc]) * (1.0f / 3.0f);
    for (int pi = threadIdx.x >> 6; pi < PTILE; pi += 4) {
        float acc = bias;
        const float* wp = wcomb + (size_t)oc * 128 * 15;
        for (int ic = 0; ic < 128; ++ic) {
            const float* xr = tile + ic * W + pi;
            #pragma unroll
            for (int t = 0; t < 15; ++t) acc += wp[ic * 15 + t] * xr[t];
        }
        y[((size_t)(b * PADL + p0 + pi)) * 64 + oc] = acc;
    }
}

// emb_seq[b][c] = mean_p (out_cnn[b][p][c] * mask[b][p])   (mean over 512)
__global__ void k_emb_seq(const float* __restrict__ cnn, const float* __restrict__ mask,
                          int Bn, float* __restrict__ out) {
    int i = blockIdx.x * blockDim.x + threadIdx.x;
    if (i >= Bn * HS) return;
    int b = i / HS, c = i % HS;
    float s = 0.f;
    for (int p = 0; p < PADL; ++p)
        s += cnn[((size_t)b * PADL + p) * HS + c] * mask[b * PADL + p];
    out[i] = s * (1.0f / PADL);
}

// ---------------------------------------------------------------------------
// Host side
// ---------------------------------------------------------------------------
static void gemm(const __bf16* A, const __bf16* Bt, const float* bias, float* C,
                 int M, int Nc, int K, int relu, hipStream_t s) {
    int tiles = (Nc >> 6) * (M >> 4);
    dim3 g((tiles + GEMM_WAVES - 1) / GEMM_WAVES);
    k_gemm_bf16<<<g, 256, 0, s>>>(A, Bt, bias, C, M, Nc, K, relu);
}

static void run_gat(const int* src, const int* dst, int E, int Etot, int Nn,
                    const float* h, const float* aw_s, const float* aw_d, const float* bias,
                    int H, int C, float* As, float* Ad, unsigned* MORD, float* DEN,
                    float* EBUF, float* AGG, __bf16* bfout, float* fout, hipStream_t s) {
    k_attn_coeff<<<cdivz((size_t)Nn * H, 256), 256, 0, s>>>(h, aw_s, aw_d, As, Ad, Nn, H, C);
    k_fill_u32<<<cdivz((size_t)Nn * H, 256), 256, 0, s>>>(MORD, 0x007FFFFFu, (size_t)Nn * H);
    k_fill_u32<<<cdivz((size_t)Nn * H, 256), 256, 0, s>>>((unsigned*)DEN, 0u, (size_t)Nn * H);
    k_fill_u32<<<cdivz((size_t)Nn * H * C, 256), 256, 0, s>>>((unsigned*)AGG, 0u, (size_t)Nn * H * C);
    k_edge_logit<<<cdivz((size_t)Etot * H, 256), 256, 0, s>>>(src, dst, E, Etot, As, Ad, EBUF, MORD, H);
    k_edge_exp<<<cdivz((size_t)Etot * H, 256), 256, 0, s>>>(dst, E, Etot, EBUF, MORD, DEN, H);
    k_edge_agg<<<cdivz((size_t)Etot * H * (C / 4), 256), 256, 0, s>>>(src, dst, E, Etot, EBUF, DEN, h, AGG, H, C);
    k_head_mean<<<cdivz((size_t)Nn * C, 256), 256, 0, s>>>(AGG, bias, Nn, H, C, bfout, fout);
}

extern "C" void kernel_launch(void* const* d_in, const int* in_sizes, int n_in,
                              void* d_out, int out_size, void* d_ws, size_t ws_size,
                              hipStream_t stream) {
    // ---- inputs (setup_inputs dict order) ----
    const int*   x        = (const int*)  d_in[0];
    const int*   ei       = (const int*)  d_in[1];
    const float* emb      = (const float*)d_in[2];
    const int*   node_len = (const int*)  d_in[3];
    // d_in[4] = batch (unused: contiguous segments derived from node_len)
    const float* tab_r    = (const float*)d_in[5];
    const float* tab_g    = (const float*)d_in[6];
    const float* W1  = (const float*)d_in[7];
    const float* a1s = (const float*)d_in[8];
    const float* a1d = (const float*)d_in[9];
    const float* b1  = (const float*)d_in[10];
    const float* W2  = (const float*)d_in[11];
    const float* a2s = (const float*)d_in[12];
    const float* a2d = (const float*)d_in[13];
    const float* b2  = (const float*)d_in[14];
    const float* W3  = (const float*)d_in[15];
    const float* a3s = (const float*)d_in[16];
    const float* a3d = (const float*)d_in[17];
    const float* b3  = (const float*)d_in[18];
    const float* leW = (const float*)d_in[19];
    const float* leb = (const float*)d_in[20];
    const float* cw1 = (const float*)d_in[21];
    const float* cb1 = (const float*)d_in[22];
    const float* cw2 = (const float*)d_in[23];
    const float* cb2 = (const float*)d_in[24];
    const float* cw3 = (const float*)d_in[25];
    const float* cb3 = (const float*)d_in[26];
    const float* l1W = (const float*)d_in[27];
    const float* l1b = (const float*)d_in[28];
    const float* l2W = (const float*)d_in[29];
    const float* l2b = (const float*)d_in[30];

    const int Nn   = in_sizes[0];          // 102400
    const int E    = in_sizes[1] / 2;      // 819200
    const int Bn   = in_sizes[3];          // 256
    const int Etot = E + Nn;               // + self loops
    const int* src = ei;
    const int* dst = ei + E;

    const size_t MSEQ = (size_t)Bn * PADL; // 131072 rows of the sequence GEMMs

    // ---- workspace bump allocator ----
    char* w = (char*)d_ws;
    auto alloc = [&](size_t bytes) -> void* {
        void* p = (void*)w;
        w += (bytes + 255) & ~(size_t)255;
        return p;
    };
    __bf16* W1t  = (__bf16*)alloc((size_t)1024 * 128 * 2);
    __bf16* W2t  = (__bf16*)alloc((size_t)1024 * 256 * 2);
    __bf16* W3t  = (__bf16*)alloc((size_t)128 * 256 * 2);
    __bf16* leWt = (__bf16*)alloc((size_t)128 * 640 * 2);
    __bf16* l1Wt = (__bf16*)alloc((size_t)512 * 64 * 2);
    __bf16* l2Wt = (__bf16*)alloc((size_t)128 * 512 * 2);
    __bf16* ABF  = (__bf16*)alloc((size_t)Nn * 1024 * 2);   // reusable bf16 A buffer
    float*  BIG1 = (float*) alloc((size_t)Nn * 1024 * 4);   // GEMM outputs h / hidden / xt
    float*  BIG2 = (float*) alloc((size_t)Nn * 1024 * 4);   // aggregation target / conv y
    float*  HFIN = (float*) alloc((size_t)Nn * HS * 4);     // final node features
    float*  EMB2 = (float*) alloc((size_t)Nn * HS * 4);     // relu(emb@le_W+le_b)
    float*  EBUF = (float*) alloc((size_t)Etot * 4 * 4);    // per-edge logits/exp
    float*  AS   = (float*) alloc((size_t)Nn * 4 * 4);
    float*  AD   = (float*) alloc((size_t)Nn * 4 * 4);
    unsigned* MORD = (unsigned*)alloc((size_t)Nn * 4 * 4);
    float*  DEN  = (float*) alloc((size_t)Nn * 4 * 4);
    int*    OFFS = (int*)   alloc((size_t)(Bn + 1) * 4);
    float*  WCOMB= (float*) alloc((size_t)64 * 128 * 15 * 4);

    // ---- output layout (flat concat in return order) ----
    float* o_cnn   = (float*)d_out;                          // [B,512,128]
    float* o_graph = o_cnn   + MSEQ * HS;                    // [B,512,128]
    float* o_mask1 = o_graph + MSEQ * HS;                    // [B,512]
    float* o_mask2 = o_mask1 + MSEQ;                         // [B,512]
    float* o_eseq  = o_mask2 + MSEQ;                         // [B,128]
    float* o_egra  = o_eseq  + (size_t)Bn * HS;              // [B,128]

    // ---- weight prep (bf16, N-major transpose) ----
    auto tc = [&](const float* Wp, __bf16* Bt, int K, int Nc) {
        k_transpose_cast<<<cdivz((size_t)K * Nc, 256), 256, 0, stream>>>(Wp, Bt, K, Nc);
    };
    tc(W1, W1t, 128, 1024);
    tc(W2, W2t, 256, 1024);
    tc(W3, W3t, 256, 128);
    tc(leW, leWt, 640, 128);
    tc(l1W, l1Wt, 64, 512);
    tc(l2W, l2Wt, 512, 128);
    k_offsets<<<1, 32, 0, stream>>>(node_len, OFFS, Bn);
    k_wcomb<<<cdivz(64 * 128 * 15, 256), 256, 0, stream>>>(cw1, cw2, cw3, WCOMB);

    // ---- GAT layer 1: x_g @ W1 (K=128 -> 4x256) ----
    k_gather_g<<<cdivz((size_t)Nn * HS, 256), 256, 0, stream>>>(x, tab_g, ABF, Nn);
    gemm(ABF, W1t, nullptr, BIG1, Nn, 1024, 128, 0, stream);
    run_gat(src, dst, E, Etot, Nn, BIG1, a1s, a1d, b1, 4, HD,
            AS, AD, MORD, DEN, EBUF, BIG2, ABF, nullptr, stream);

    // ---- GAT layer 2: act1 @ W2 (K=256 -> 4x256) ----
    gemm(ABF, W2t, nullptr, BIG1, Nn, 1024, 256, 0, stream);
    run_gat(src, dst, E, Etot, Nn, BIG1, a2s, a2d, b2, 4, HD,
            AS, AD, MORD, DEN, EBUF, BIG2, ABF, nullptr, stream);

    // ---- GAT layer 3: act2 @ W3 (K=256 -> 1x128) ----
    gemm(ABF, W3t, nullptr, BIG1, Nn, 128, 256, 0, stream);
    run_gat(src, dst, E, Etot, Nn, BIG1, a3s, a3d, b3, 1, HS,
            AS, AD, MORD, DEN, EBUF, BIG2, nullptr, HFIN, stream);

    // ---- graph pooling ----
    k_graph_pool<<<cdivz((size_t)Bn * HS, 256), 256, 0, stream>>>(HFIN, OFFS, node_len, Bn, o_egra);

    // ---- emb2 = relu(emb @ le_W + le_b) ----
    k_cast_bf16<<<cdivz((size_t)Nn * 640, 256), 256, 0, stream>>>(emb, ABF, (size_t)Nn * 640);
    gemm(ABF, leWt, leb, EMB2, Nn, 128, 640, 1, stream);

    // ---- masks, out_graph ----
    k_mask<<<cdivz((size_t)Bn * PADL, 256), 256, 0, stream>>>(node_len, o_mask1, o_mask2, Bn);
    k_outgraph<<<cdivz(MSEQ * HS, 256), 256, 0, stream>>>(HFIN, OFFS, node_len, Nn, Bn, o_graph);

    // ---- out_r transposed into xt [B,128,512] (BIG1 reuse) ----
    k_build_xt<<<cdivz((size_t)Bn * HS * PADL, 256), 256, 0, stream>>>(
        x, tab_r, EMB2, OFFS, node_len, Nn, Bn, BIG1);

    // ---- fused conv (7/11/15 taps) -> y [B*512, 64] (BIG2 reuse) ----
    {
        dim3 g(Bn, PADL / PTILE);
        k_conv<<<g, 256, 0, stream>>>(BIG1, WCOMB, cb1, cb2, cb3, BIG2);
    }

    // ---- MLP: relu(y @ l1W + l1b) @ l2W + l2b -> out_seq_cnn ----
    k_cast_bf16<<<cdivz(MSEQ * 64, 256), 256, 0, stream>>>(BIG2, ABF, MSEQ * 64);
    gemm(ABF, l1Wt, l1b, BIG1, (int)MSEQ, 512, 64, 1, stream);
    k_cast_bf16<<<cdivz(MSEQ * 512, 256), 256, 0, stream>>>(BIG1, ABF, MSEQ * 512);
    gemm(ABF, l2Wt, l2b, o_cnn, (int)MSEQ, 128, 512, 0, stream);

    // ---- emb_seq = masked mean over positions ----
    k_emb_seq<<<cdivz((size_t)Bn * HS, 256), 256, 0, stream>>>(o_cnn, o_mask1, Bn, o_eseq);

    (void)n_in; (void)out_size; (void)ws_size;
}
